// SmyrfAttention_34583076667862
// MI455X (gfx1250) — matrix operand order repaired and checked by reading, compiled, not run on hardware
//
#include <hip/hip_runtime.h>
#include <hip/hip_bf16.h>
#include <math.h>

#define N_HASHES 8
#define BATCH    4
#define SEQ      8192
#define DIM      64
#define DV       64
#define QATTN    128
#define NCLUST   (SEQ / QATTN)   /* 64 */

#define QROW 80    /* Q/K LDS row stride in halves: 160B = 5*32B  */
#define VROW 144   /* Vt/Ps LDS row stride in halves: 288B = 9*32B */

typedef __attribute__((ext_vector_type(16))) _Float16 v16h;
typedef __attribute__((ext_vector_type(8)))  _Float16 v8h;
typedef __attribute__((ext_vector_type(8)))  float    v8f;

// ---------------------------------------------------------------------------
// 0) deterministic pseudo-random LSH parameters (alpha ~ N(0,1), beta ~ U[0,1))
// ---------------------------------------------------------------------------
__device__ __forceinline__ unsigned pcg_hash(unsigned x) {
  x = x * 747796405u + 2891336453u;
  unsigned w = ((x >> ((x >> 28u) + 4u)) ^ x) * 277803737u;
  return (w >> 22u) ^ w;
}

__global__ void gen_hash_kernel(float* __restrict__ alpha, float* __restrict__ beta) {
  int t = threadIdx.x;
  if (t < N_HASHES * 66) {
    unsigned u1 = pcg_hash(2u * t + 1u);
    unsigned u2 = pcg_hash(2u * t + 2u);
    float f1 = (float)(u1 | 1u) * 2.3283064365386963e-10f;   // (0,1]
    float f2 = (float)u2 * 2.3283064365386963e-10f;          // [0,1)
    alpha[t] = sqrtf(-2.0f * __logf(f1)) * __cosf(6.28318530717958647f * f2);
  }
  if (t < N_HASHES) {
    beta[t] = (float)pcg_hash(99991u + t) * 2.3283064365386963e-10f;
  }
}

// ---------------------------------------------------------------------------
// 1) per-batch max squared norms for the XBOX+ transform
// ---------------------------------------------------------------------------
__global__ __launch_bounds__(256) void maxnorm_kernel(const float* __restrict__ q,
                                                      const float* __restrict__ k,
                                                      float* __restrict__ M2arr) {
  int b = blockIdx.x, sel = blockIdx.y;
  const float* src = sel ? k : q;
  float mx = 0.0f;
  for (int n = threadIdx.x; n < SEQ; n += 256) {
    const float* row = src + ((long)b * SEQ + n) * DIM;
    float s = 0.0f;
#pragma unroll
    for (int d = 0; d < DIM; d += 4) {
      float4 x = *(const float4*)(row + d);
      s += x.x * x.x + x.y * x.y + x.z * x.z + x.w * x.w;
    }
    mx = fmaxf(mx, s);
  }
  __shared__ float red[256];
  red[threadIdx.x] = mx;
  __syncthreads();
  for (int s = 128; s > 0; s >>= 1) {
    if (threadIdx.x < s) red[threadIdx.x] = fmaxf(red[threadIdx.x], red[threadIdx.x + s]);
    __syncthreads();
  }
  if (threadIdx.x == 0) M2arr[sel * BATCH + b] = red[0];
}

// ---------------------------------------------------------------------------
// 2) E2LSH bucket id per (hash, batch, token) for q (sel=0) and k (sel=1)
// ---------------------------------------------------------------------------
__global__ __launch_bounds__(256) void bucket_kernel(const float* __restrict__ q,
                                                     const float* __restrict__ k,
                                                     const float* __restrict__ alpha,
                                                     const float* __restrict__ beta,
                                                     const float* __restrict__ M2arr,
                                                     float* __restrict__ qb,
                                                     float* __restrict__ kb) {
  long tid = (long)blockIdx.x * 256 + threadIdx.x;
  const long total = (long)N_HASHES * BATCH * SEQ;
  int sel = (int)(tid / total);
  long r = tid % total;
  int h = (int)(r / ((long)BATCH * SEQ));
  int b = (int)((r / SEQ) % BATCH);
  int n = (int)(r % SEQ);
  const float* src = sel ? k : q;
  const float* row = src + ((long)b * SEQ + n) * DIM;
  const float* a = alpha + h * 66;
  float norm2 = 0.0f, dot = 0.0f;
#pragma unroll 8
  for (int d = 0; d < DIM; ++d) {
    float x = row[d];
    norm2 += x * x;
    dot += x * a[d];
  }
  float M2 = M2arr[b] + M2arr[BATCH + b];
  float ext = sqrtf(fmaxf(M2 - norm2, 0.0f));
  dot += ext * a[64 + sel];                 // Q(x) uses slot 64, K(x) uses slot 65
  float bkt = floorf(dot + beta[h]);        // LSH_R == 1.0
  (sel ? kb : qb)[r] = bkt;
}

// ---------------------------------------------------------------------------
// 3) per-(tensor,hash,batch) argsort of 8192 bucket ids: bitonic sort in LDS
// ---------------------------------------------------------------------------
__global__ __launch_bounds__(1024) void sort_kernel(const float* __restrict__ qb,
                                                    const float* __restrict__ kb,
                                                    int* __restrict__ qpos,
                                                    int* __restrict__ kpos) {
  __shared__ unsigned long long key[SEQ];   // 64 KB
  int blk = blockIdx.x;                     // 0 .. 2*H*B-1
  int sel = blk / (N_HASHES * BATCH);
  int hb  = blk % (N_HASHES * BATCH);
  const float* bkt = (sel ? kb : qb) + (long)hb * SEQ;
  int* pos = (sel ? kpos : qpos) + (long)hb * SEQ;

  for (int i = threadIdx.x; i < SEQ; i += 1024) {
    unsigned u = __float_as_uint(bkt[i]);
    u = (u & 0x80000000u) ? ~u : (u | 0x80000000u);   // order-preserving float->uint
    key[i] = ((unsigned long long)u << 32) | (unsigned)i;
  }
  __syncthreads();

  for (int kk = 2; kk <= SEQ; kk <<= 1) {
    for (int j = kk >> 1; j > 0; j >>= 1) {
      for (int i = threadIdx.x; i < SEQ; i += 1024) {
        int ixj = i ^ j;
        if (ixj > i) {
          bool up = ((i & kk) == 0);
          unsigned long long a = key[i], c = key[ixj];
          if ((a > c) == up) { key[i] = c; key[ixj] = a; }
        }
      }
      __syncthreads();
    }
  }
  for (int i = threadIdx.x; i < SEQ; i += 1024)
    pos[i] = (int)(key[i] & 0xFFFFFFFFu);
}

// ---------------------------------------------------------------------------
// 4) clustered attention: one 256-thread block per (h, b, cluster)
//    QK^T and P*V via v_wmma_f32_16x16x32_f16, fragments staged in LDS
// ---------------------------------------------------------------------------
// A-fragment (16-bit A 16x32, ISA 7.12.2): lane half 'lh' selects K base,
// halves[0..7] = K{o..o+7}, halves[8..15] = K{o+16..o+23}.
__device__ __forceinline__ v16h load_a_frag(const _Float16* rowp, int o) {
  union { struct { v8h lo, hi; } s; v16h v; } u;
  u.s.lo = *(const v8h*)(rowp + o);
  u.s.hi = *(const v8h*)(rowp + o + 16);
  return u.v;
}

__global__ __launch_bounds__(256) void attn_kernel(const float* __restrict__ q,
                                                   const float* __restrict__ k,
                                                   const float* __restrict__ v,
                                                   const int* __restrict__ qpos,
                                                   const int* __restrict__ kpos,
                                                   float* __restrict__ out_ws,
                                                   float* __restrict__ lse_ws) {
  // 10240 + 10240 + 9216 halves = 59,392 bytes; Ps overlays Qs+Ks after QK^T.
  __shared__ __align__(32) _Float16 smem[2 * 128 * QROW + 64 * VROW];
  _Float16* Qs = smem;
  _Float16* Ks = smem + 128 * QROW;
  _Float16* Vt = smem + 2 * 128 * QROW;
  _Float16* Ps = smem;                       // reuse after barrier

  int bid = blockIdx.x;
  int c = bid % NCLUST;
  int b = (bid / NCLUST) % BATCH;
  int h = bid / (NCLUST * BATCH);
  long hb = (long)h * BATCH + b;
  const int* qp = qpos + hb * SEQ + c * QATTN;
  const int* kp = kpos + hb * SEQ + c * QATTN;
  int tid = threadIdx.x;

  // ---- stage gathered tiles into LDS (fp32 -> fp16), V transposed -------
  for (int i = tid; i < 128 * 16; i += 256) {
    int r = i >> 4;
    int c4 = (i & 15) * 4;
    int qr = qp[r];
    int kr = kp[r];
    float4 xq = *(const float4*)(q + ((long)b * SEQ + qr) * DIM + c4);
    float4 xk = *(const float4*)(k + ((long)b * SEQ + kr) * DIM + c4);
    float4 xv = *(const float4*)(v + ((long)b * SEQ + kr) * DIM + c4);
    _Float16* pq = &Qs[r * QROW + c4];
    pq[0] = (_Float16)xq.x; pq[1] = (_Float16)xq.y;
    pq[2] = (_Float16)xq.z; pq[3] = (_Float16)xq.w;
    _Float16* pk = &Ks[r * QROW + c4];
    pk[0] = (_Float16)xk.x; pk[1] = (_Float16)xk.y;
    pk[2] = (_Float16)xk.z; pk[3] = (_Float16)xk.w;
    Vt[(c4 + 0) * VROW + r] = (_Float16)xv.x;
    Vt[(c4 + 1) * VROW + r] = (_Float16)xv.y;
    Vt[(c4 + 2) * VROW + r] = (_Float16)xv.z;
    Vt[(c4 + 3) * VROW + r] = (_Float16)xv.w;
  }
  __syncthreads();

  int w  = tid >> 5;         // wave id, owns rows w*16 .. w*16+15
  int l  = tid & 31;
  int lh = l >> 4;           // half-wave select
  int ll = l & 15;

  // ---- S = Q K^T : 2 K-steps (D=64) x 8 N-tiles ------------------------
  const _Float16* qrow = &Qs[(w * 16 + ll) * QROW];
  v8f acc[8] = {};
#pragma unroll
  for (int kk = 0; kk < 2; ++kk) {
    v16h a = load_a_frag(qrow, kk * 32 + lh * 8);
#pragma unroll
    for (int j = 0; j < 8; ++j) {
      const _Float16* krow = &Ks[(j * 16 + ll) * QROW];
      v16h bf = load_a_frag(krow, kk * 32 + lh * 8);
      acc[j] = __builtin_amdgcn_wmma_f32_16x16x32_f16(
          false, a, false, bf, (short)0, acc[j], false, false);
    }
  }

  // ---- softmax over 128 keys per row (row split across 16 lanes) -------
  float srow[8], lsr[8];
#pragma unroll
  for (int r = 0; r < 8; ++r) {
    float m = acc[0][r];
#pragma unroll
    for (int j = 1; j < 8; ++j) m = fmaxf(m, acc[j][r]);
#pragma unroll
    for (int off = 1; off < 16; off <<= 1) m = fmaxf(m, __shfl_xor(m, off, 32));
    float s = 0.0f;
#pragma unroll
    for (int j = 0; j < 8; ++j) {
      float e = __expf(acc[j][r] - m);
      acc[j][r] = e;
      s += e;
    }
#pragma unroll
    for (int off = 1; off < 16; off <<= 1) s += __shfl_xor(s, off, 32);
    srow[r] = s;
    lsr[r]  = m + __logf(s);
  }
  __syncthreads();   // all Qs/Ks reads done -> safe to overlay Ps

  // ---- spill P (unnormalized probs) to LDS as fp16 ---------------------
#pragma unroll
  for (int r = 0; r < 8; ++r) {
    int row = w * 16 + lh * 8 + r;
#pragma unroll
    for (int j = 0; j < 8; ++j)
      Ps[row * VROW + j * 16 + ll] = (_Float16)acc[j][r];
  }
  __syncthreads();

  // ---- O = P V : 4 K-steps (128 keys) x 4 N-tiles (Dv=64) --------------
  v8f oacc[4] = {};
  const _Float16* prow = &Ps[(w * 16 + ll) * VROW];
#pragma unroll
  for (int kk = 0; kk < 4; ++kk) {
    v16h a = load_a_frag(prow, kk * 32 + lh * 8);
#pragma unroll
    for (int j = 0; j < 4; ++j) {
      const _Float16* vrow = &Vt[(j * 16 + ll) * VROW];
      v16h bf = load_a_frag(vrow, kk * 32 + lh * 8);
      oacc[j] = __builtin_amdgcn_wmma_f32_16x16x32_f16(
          false, a, false, bf, (short)0, oacc[j], false, false);
    }
  }

  // ---- normalize + scatter back to original query order ----------------
#pragma unroll
  for (int r = 0; r < 8; ++r) {
    int row = w * 16 + lh * 8 + r;
    int orig = qp[row];
    float rinv = 1.0f / srow[r];
    float* orow = out_ws + (hb * SEQ + orig) * DV;
#pragma unroll
    for (int j = 0; j < 4; ++j)
      orow[j * 16 + ll] = oacc[j][r] * rinv;
    if (ll == 0) lse_ws[hb * SEQ + orig] = lsr[r];
  }
}

// ---------------------------------------------------------------------------
// 5) combine hash rounds with softmax over per-row logsumexp
// ---------------------------------------------------------------------------
__global__ __launch_bounds__(256) void combine_kernel(const float* __restrict__ out_ws,
                                                      const float* __restrict__ lse_ws,
                                                      float* __restrict__ out) {
  long t = (long)blockIdx.x * 256 + threadIdx.x;   // B*SEQ*16 threads
  int dv4 = (int)(t & 15);
  long bq = t >> 4;
  const long stride = (long)BATCH * SEQ;
  float lse[N_HASHES];
  float m = -INFINITY;
#pragma unroll
  for (int h = 0; h < N_HASHES; ++h) {
    lse[h] = lse_ws[(long)h * stride + bq];
    m = fmaxf(m, lse[h]);
  }
  float s = 0.0f, wgt[N_HASHES];
#pragma unroll
  for (int h = 0; h < N_HASHES; ++h) { wgt[h] = __expf(lse[h] - m); s += wgt[h]; }
  float inv = 1.0f / s;
  float4 a = {0.0f, 0.0f, 0.0f, 0.0f};
#pragma unroll
  for (int h = 0; h < N_HASHES; ++h) {
    float4 x = *(const float4*)(out_ws + ((long)h * stride + bq) * DV + dv4 * 4);
    float wv = wgt[h] * inv;
    a.x += wv * x.x; a.y += wv * x.y; a.z += wv * x.z; a.w += wv * x.w;
  }
  *(float4*)(out + bq * DV + dv4 * 4) = a;
}

// ---------------------------------------------------------------------------
// launch
// ---------------------------------------------------------------------------
extern "C" void kernel_launch(void* const* d_in, const int* in_sizes, int n_in,
                              void* d_out, int out_size, void* d_ws, size_t ws_size,
                              hipStream_t stream) {
  const float* q = (const float*)d_in[0];
  const float* k = (const float*)d_in[1];
  const float* v = (const float*)d_in[2];
  float* out = (float*)d_out;

  char* p = (char*)d_ws;
  auto alloc = [&](size_t bytes) -> void* {
    void* r = (void*)p;
    p += (bytes + 255) & ~(size_t)255;
    return r;
  };
  const size_t HBN = (size_t)N_HASHES * BATCH * SEQ;
  float* alpha = (float*)alloc(N_HASHES * 66 * sizeof(float));
  float* beta  = (float*)alloc(N_HASHES * sizeof(float));
  float* M2    = (float*)alloc(2 * BATCH * sizeof(float));
  float* qb    = (float*)alloc(HBN * sizeof(float));
  float* kb    = (float*)alloc(HBN * sizeof(float));
  int*   qpos  = (int*)alloc(HBN * sizeof(int));
  int*   kpos  = (int*)alloc(HBN * sizeof(int));
  float* lse   = (float*)alloc(HBN * sizeof(float));
  float* ows   = (float*)alloc(HBN * (size_t)DV * sizeof(float));

  gen_hash_kernel<<<1, 544, 0, stream>>>(alpha, beta);
  maxnorm_kernel<<<dim3(BATCH, 2), 256, 0, stream>>>(q, k, M2);
  {
    long total = 2L * N_HASHES * BATCH * SEQ;
    bucket_kernel<<<(unsigned)((total + 255) / 256), 256, 0, stream>>>(
        q, k, alpha, beta, M2, qb, kb);
  }
  sort_kernel<<<2 * N_HASHES * BATCH, 1024, 0, stream>>>(qb, kb, qpos, kpos);
  attn_kernel<<<N_HASHES * BATCH * NCLUST, 256, 0, stream>>>(
      q, k, v, qpos, kpos, ows, lse);
  {
    long total = (long)BATCH * SEQ * 16;
    combine_kernel<<<(unsigned)(total / 256), 256, 0, stream>>>(ows, lse, out);
  }
}
